// AlignDeformConv2d_77824807403838
// MI455X (gfx1250) — matrix operand order repaired and checked by reading, not compiled, not run on hardware
//
#include <hip/hip_runtime.h>
#include <hip/hip_bf16.h>

typedef _Float16 v16h __attribute__((ext_vector_type(16)));
typedef _Float16 v8h  __attribute__((ext_vector_type(8)));
typedef float    v8f  __attribute__((ext_vector_type(8)));
// under-aligned float2 (4B) so clamped-base pair loads are legal at any dword address
typedef float    f2v  __attribute__((ext_vector_type(2))) __attribute__((aligned(4)));

static constexpr int BATCHN = 8;
static constexpr int CMEM   = 64;
static constexpr int CQUE   = 64;
static constexpr int COUT   = 64;
static constexpr int HT     = 128;
static constexpr int WD     = 128;
static constexpr int HWC    = HT * WD;           // 16384
static constexpr int KS     = 3;
static constexpr int KTAP   = 9;                 // 3x3 taps
static constexpr int CIN1   = CMEM + CQUE;       // 128
static constexpr int K1     = CIN1 * KTAP;       // 1152
static constexpr int K2     = CMEM * KTAP;       // 576
static constexpr int NPIX   = BATCHN * HWC;      // 131072
static constexpr int TILE_P = 32;
static constexpr int NTILES = NPIX / TILE_P;     // 4096
// LDS S-tile row stride in halves: 584 halves = 292 dwords = 4*73 dwords.
// 16B-aligned rows; 73 coprime with 16 -> 16B lane groups hit distinct banks.
static constexpr int SROW   = 584;

// -------------------------------------------------------------------------
// Pack weights to f16 GEMM-A layouts:
//   wA1[32][1152]: row o (o>=27 zero), col ck = ci*9 + tap       (offset conv)
//   wA2[64][576] : row o,             col ck = k*64 + c          (deform einsum)
// -------------------------------------------------------------------------
__global__ __launch_bounds__(256) void pack_weights(const float* __restrict__ w_off,
                                                    const float* __restrict__ w_def,
                                                    _Float16* __restrict__ wA1,
                                                    _Float16* __restrict__ wA2) {
    int idx = blockIdx.x * 256 + threadIdx.x;
    if (idx < 32 * K1) {
        int o = idx / K1, ck = idx % K1;
        float v = 0.f;
        if (o < 27) {
            int ci = ck / KTAP, tap = ck % KTAP;
            v = w_off[(o * CIN1 + ci) * KTAP + tap];
        }
        wA1[idx] = (_Float16)v;
    } else {
        int j = idx - 32 * K1;
        if (j < COUT * K2) {
            int o = j / K2, ck = j % K2;
            int k = ck / CMEM, c = ck % CMEM;
            wA2[j] = (_Float16)w_def[(o * CMEM + c) * KTAP + k];
        }
    }
}

// -------------------------------------------------------------------------
// Kernel 1: offset conv as implicit-GEMM.  D[o,p] = sum_ck wA1[o,ck]*im2col[ck,p]
// One workgroup = 32 consecutive pixels in one row (b,h uniform). 128 threads,
// 4 waves = 2 o-tiles x 2 p-tiles. K=1152 in two 576 phases; phase 0 == mem
// channels, phase 1 == que channels (uniform base pointer, no per-elem select).
// -------------------------------------------------------------------------
__global__ __launch_bounds__(128) void offset_conv_wmma(const float* __restrict__ mem,
                                                        const float* __restrict__ que,
                                                        const _Float16* __restrict__ wA1,
                                                        const float* __restrict__ b_off,
                                                        float* __restrict__ offw) {
    __shared__ _Float16 S[TILE_P * SROW];
    const int tid    = threadIdx.x;
    const int pixel0 = blockIdx.x * TILE_P;
    const int b      = pixel0 >> 14;
    const int pix    = pixel0 & (HWC - 1);
    const int h      = pix >> 7;
    const int w0     = pix & (WD - 1);

    const int p     = tid & 31;          // pixel within tile (stagers)
    const int rslot = tid >> 5;          // 0..3
    const int xpix  = w0 + p;

    const int wave = tid >> 5;
    const int lane = tid & 31;
    const int hi   = lane >> 4;          // upper half-wave
    const int nm   = lane & 15;          // M (for A) / N (for B,D)
    const int ot   = wave & 1;
    const int pt   = wave >> 1;
    const _Float16* Arow = wA1 + (ot * 16 + nm) * K1;
    const _Float16* Brow = S + (pt * 16 + nm) * SROW;

    v8f acc = {0.f, 0.f, 0.f, 0.f, 0.f, 0.f, 0.f, 0.f};

    for (int phase = 0; phase < 2; ++phase) {
        // ---- stage im2col rows for 64 channels (cc) x 9 taps, S[p][cc*9+tap] ----
        const float* xbase = (phase == 0 ? mem : que) + ((b * CMEM) << 14);
        for (int cc = rslot; cc < CMEM; cc += 4) {
            const float* src = xbase + (cc << 14);
            _Float16* srow = S + p * SROW + cc * KTAP;
#pragma unroll
            for (int tap = 0; tap < KTAP; ++tap) {
                const int yy = h + tap / KS - 1;
                const int xx = xpix + tap % KS - 1;
                float v = 0.f;
                if ((unsigned)yy < (unsigned)HT && (unsigned)xx < (unsigned)WD)
                    v = src[(yy << 7) + xx];
                srow[tap] = (_Float16)v;
            }
        }
        __syncthreads();

        // ---- 18 WMMAs over this phase ----
        const _Float16* Aph = Arow + phase * K2;
        for (int kk = 0; kk < K2 / 32; ++kk) {
            const int ka = kk * 32 + hi * 8;
            v8h a0 = *(const v8h*)(Aph + ka);
            v8h a1 = *(const v8h*)(Aph + ka + 16);
            const int kb = kk * 32 + hi * 8;
            v8h b0 = *(const v8h*)(Brow + kb);
            v8h b1 = *(const v8h*)(Brow + kb + 16);
            v16h Af = __builtin_shufflevector(a0, a1, 0,1,2,3,4,5,6,7,8,9,10,11,12,13,14,15);
            v16h Bf = __builtin_shufflevector(b0, b1, 0,1,2,3,4,5,6,7,8,9,10,11,12,13,14,15);
            acc = __builtin_amdgcn_wmma_f32_16x16x32_f16(false, Af, false, Bf,
                                                         (short)0, acc, false, false);
        }
        __syncthreads();   // protect S before next phase's staging
    }

    // ---- store 27 values per pixel (+bias), pixel-major stride 32 ----
    const int pixel = pixel0 + pt * 16 + nm;
    float* orow = offw + pixel * 32;
#pragma unroll
    for (int v = 0; v < 8; ++v) {
        const int o = ot * 16 + hi * 8 + v;
        if (o < 27) orow[o] = acc[v] + b_off[o];
    }
}

// -------------------------------------------------------------------------
// Kernel 2: bilinear sampling (fused) + einsum GEMM.
// out[o,p] = sum_{k,c} wA2[o, k*64+c] * sample[k,c,p].
// 256 threads = 8 waves = 4 o-tiles x 2 p-tiles of 16x16.
// Geometry for all 9 taps precomputed into LDS with boundary handling folded
// into element-swizzled weights (u00..u11 apply to .x/.y of clamped pair loads).
// -------------------------------------------------------------------------
__global__ __launch_bounds__(256) void deform_gemm_wmma(const float* __restrict__ mem,
                                                        const float* __restrict__ offw,
                                                        const _Float16* __restrict__ wA2,
                                                        float* __restrict__ out) {
    __shared__ _Float16 S[TILE_P * SROW];
    __shared__ int   gRO0[KTAP][TILE_P];     // clamped row offset (y0c<<7)
    __shared__ int   gRO1[KTAP][TILE_P];     // clamped row offset (y1c<<7)
    __shared__ int   gXB [KTAP][TILE_P];     // clamped pair base x in [0,126]
    __shared__ float gU  [KTAP][TILE_P][4];  // weights for {row0.x,row0.y,row1.x,row1.y}

    const int tid    = threadIdx.x;
    const int pixel0 = blockIdx.x * TILE_P;
    const int b      = pixel0 >> 14;
    const int pix    = pixel0 & (HWC - 1);
    const int h      = pix >> 7;
    const int w0     = pix & (WD - 1);
    const float* membase = mem + ((b * CMEM) << 14);

    // ---- geometry for all (tap, pixel) tasks: 288 tasks over 256 threads ----
    for (int t = tid; t < KTAP * TILE_P; t += 256) {
        const int k  = t >> 5;
        const int pp = t & 31;
        const float* o27 = offw + (pixel0 + pp) * 32;
        const float dy = o27[2 * k];
        const float dx = o27[2 * k + 1];
        const float mk = o27[18 + k];
        const float py = (float)(h + k / KS - 1) + dy;
        const float px = (float)(w0 + pp + k % KS - 1) + dx;
        const float y0f = floorf(py), x0f = floorf(px);
        const float fy = py - y0f, fx = px - x0f;
        const int y0 = (int)y0f, x0 = (int)x0f;
        const float vy0 = ((unsigned)y0       < (unsigned)HT) ? 1.f : 0.f;
        const float vy1 = ((unsigned)(y0 + 1) < (unsigned)HT) ? 1.f : 0.f;
        const float vx0 = ((unsigned)x0       < (unsigned)WD) ? 1.f : 0.f;
        const float vx1 = ((unsigned)(x0 + 1) < (unsigned)WD) ? 1.f : 0.f;
        const float w00 = (1.f - fy) * (1.f - fx) * vy0 * vx0 * mk;
        const float w01 = (1.f - fy) * fx         * vy0 * vx1 * mk;
        const float w10 = fy         * (1.f - fx) * vy1 * vx0 * mk;
        const float w11 = fy         * fx         * vy1 * vx1 * mk;
        const int y0c = min(max(y0, 0), HT - 1);
        const int y1c = min(max(y0 + 1, 0), HT - 1);
        const int x0c = min(max(x0, 0), WD - 1);
        const int x1c = min(max(x0 + 1, 0), WD - 1);
        const int xb  = min(max(x0, 0), WD - 2);     // pair base: loads [xb, xb+1]
        const bool s0 = (x0c != xb);                 // x0 sample comes from .y
        const bool s1 = (x1c != xb);                 // x1 sample comes from .y
        gRO0[k][pp] = y0c << 7;
        gRO1[k][pp] = y1c << 7;
        gXB [k][pp] = xb;
        gU[k][pp][0] = (s0 ? 0.f : w00) + (s1 ? 0.f : w01);  // row0 .x
        gU[k][pp][1] = (s0 ? w00 : 0.f) + (s1 ? w01 : 0.f);  // row0 .y
        gU[k][pp][2] = (s0 ? 0.f : w10) + (s1 ? 0.f : w11);  // row1 .x
        gU[k][pp][3] = (s0 ? w10 : 0.f) + (s1 ? w11 : 0.f);  // row1 .y
    }
    __syncthreads();

    // ---- sampling: 2 pair-loads + 4 FMA per (tap, channel) ----
    const int p     = tid & 31;
    const int cslot = tid >> 5;          // 0..7 -> 8 channels each
    for (int k = 0; k < KTAP; ++k) {
        const int   ro0 = gRO0[k][p];
        const int   ro1 = gRO1[k][p];
        const int   xb  = gXB [k][p];
        const float u00 = gU[k][p][0], u01 = gU[k][p][1];
        const float u10 = gU[k][p][2], u11 = gU[k][p][3];
        _Float16* srow = S + p * SROW + k * CMEM;
        for (int c = cslot; c < CMEM; c += 8) {
            const float* mc = membase + (c << 14);
            f2v a = *(const f2v*)(mc + ro0 + xb);
            f2v bq = *(const f2v*)(mc + ro1 + xb);
            srow[c] = (_Float16)(u00 * a.x + u01 * a.y + u10 * bq.x + u11 * bq.y);
        }
    }
    __syncthreads();

    // ---- GEMM: 18 WMMAs per wave tile ----
    const int wave = tid >> 5;
    const int lane = tid & 31;
    const int hi   = lane >> 4;
    const int nm   = lane & 15;
    const int ot   = wave & 3;
    const int pt   = wave >> 2;
    const _Float16* Arow = wA2 + (ot * 16 + nm) * K2;
    const _Float16* Brow = S + (pt * 16 + nm) * SROW;

    v8f acc = {0.f, 0.f, 0.f, 0.f, 0.f, 0.f, 0.f, 0.f};
    for (int kk = 0; kk < K2 / 32; ++kk) {
        const int ka = kk * 32 + hi * 8;
        v8h a0 = *(const v8h*)(Arow + ka);
        v8h a1 = *(const v8h*)(Arow + ka + 16);
        const int kb = kk * 32 + hi * 8;
        v8h b0 = *(const v8h*)(Brow + kb);
        v8h b1 = *(const v8h*)(Brow + kb + 16);
        v16h Af = __builtin_shufflevector(a0, a1, 0,1,2,3,4,5,6,7,8,9,10,11,12,13,14,15);
        v16h Bf = __builtin_shufflevector(b0, b1, 0,1,2,3,4,5,6,7,8,9,10,11,12,13,14,15);
        acc = __builtin_amdgcn_wmma_f32_16x16x32_f16(false, Af, false, Bf,
                                                     (short)0, acc, false, false);
    }

    // ---- store D: out[b][o][h][w], coalesced along W within a half-wave ----
    const int pixel = pixel0 + pt * 16 + nm;
    const int ob    = pixel >> 14;
    const int opix  = pixel & (HWC - 1);
    float* obase = out + ((ob * COUT) << 14) + opix;
#pragma unroll
    for (int v = 0; v < 8; ++v) {
        const int o = ot * 16 + hi * 8 + v;
        obase[o << 14] = acc[v];
    }
}

// -------------------------------------------------------------------------
extern "C" void kernel_launch(void* const* d_in, const int* in_sizes, int n_in,
                              void* d_out, int out_size, void* d_ws, size_t ws_size,
                              hipStream_t stream) {
    (void)in_sizes; (void)n_in; (void)out_size; (void)ws_size;
    const float* mem   = (const float*)d_in[0];
    const float* que   = (const float*)d_in[1];
    const float* w_off = (const float*)d_in[2];
    const float* b_off = (const float*)d_in[3];
    const float* w_def = (const float*)d_in[4];
    float* out = (float*)d_out;

    char* ws = (char*)d_ws;
    _Float16* wA1 = (_Float16*)(ws);                                   //  73,728 B
    _Float16* wA2 = (_Float16*)(ws + (size_t)32 * K1 * 2);             //  73,728 B
    float*    offw = (float*)(ws + (size_t)32 * K1 * 2 + (size_t)COUT * K2 * 2); // 16 MiB

    pack_weights<<<(32 * K1 + COUT * K2 + 255) / 256, 256, 0, stream>>>(w_off, w_def, wA1, wA2);
    offset_conv_wmma<<<NTILES, 128, 0, stream>>>(mem, que, wA1, b_off, offw);
    deform_gemm_wmma<<<NTILES, 256, 0, stream>>>(mem, offw, wA2, out);
}
